// GTLayer_1666447311027
// MI455X (gfx1250) — matrix operand order repaired and checked by reading, not compiled, not run on hardware
//
#include <hip/hip_runtime.h>
#include <hip/hip_bf16.h>
#include <math.h>

#define NN 40000
#define EE 640000
#define CC 128
#define HH 8
#define DD 16
#define EDIM 64
#define EPSV 1e-5f

typedef __attribute__((ext_vector_type(16))) _Float16 v16h;
typedef __attribute__((ext_vector_type(8)))  float    v8f;

// ---------------------------------------------------------------------------
// WMMA GEMM:  O[M,NC] = X[M,K](f32) @ Wh[NC,K](f16)^T  (+bias)(+residual)(relu)
// One wave computes a 16 x (16*NGRP) tile: A converted once, reused NGRP times.
// K is compile-time -> fully unrolled -> NGRP * K/32 static v_wmma per site.
// Layouts used here (A and B share this k order):
//   A (16x32 f16): lane L, m=L%16, g=L/16; half j -> K=k0+16g+j (16 contiguous)
//   B (32x16 f16): lane L, n=L%16, g=L/16; half j -> K=k0+16g+j (16 contiguous)
//   C/D (16x16 f32): vgpr r -> row r+8g, col = lane%16
// ---------------------------------------------------------------------------
template<int K, int NGRP, bool BIAS, bool RES, bool RELU>
__global__ __launch_bounds__(256)
void gemm_wmma(const float* __restrict__ X, const _Float16* __restrict__ Wh,
               const float* __restrict__ bias, const float* __restrict__ residual,
               float* __restrict__ O, int M, int NC)
{
    const int lane  = threadIdx.x & 31;
    const int wave  = (blockIdx.x * blockDim.x + threadIdx.x) >> 5;
    const int nwave = (gridDim.x * blockDim.x) >> 5;
    const int mt_cnt = M >> 4;
    const int ngc    = NC / (16 * NGRP);
    const int tiles  = mt_cnt * ngc;
    const int l16 = lane & 15;
    const int g   = lane >> 4;

    for (int t = wave; t < tiles; t += nwave) {
        const int mt = t / ngc;
        const int nb = (t - mt * ngc) * NGRP;   // first 16-col tile of this wave
        const int arow = mt * 16 + l16;
        v8f acc[NGRP] = {};
        #pragma unroll
        for (int k0 = 0; k0 < K; k0 += 32) {
            v16h a;
            const float4* xa = (const float4*)(X + (size_t)arow * K + k0 + 16 * g);
            float4 t0 = xa[0], t1 = xa[1], t2 = xa[2], t3 = xa[3];
            a[0]=(_Float16)t0.x; a[1]=(_Float16)t0.y; a[2]=(_Float16)t0.z; a[3]=(_Float16)t0.w;
            a[4]=(_Float16)t1.x; a[5]=(_Float16)t1.y; a[6]=(_Float16)t1.z; a[7]=(_Float16)t1.w;
            a[8]=(_Float16)t2.x; a[9]=(_Float16)t2.y; a[10]=(_Float16)t2.z; a[11]=(_Float16)t2.w;
            a[12]=(_Float16)t3.x; a[13]=(_Float16)t3.y; a[14]=(_Float16)t3.z; a[15]=(_Float16)t3.w;
            #pragma unroll
            for (int n = 0; n < NGRP; ++n) {
                const v16h b = *(const v16h*)(Wh + (size_t)((nb + n) * 16 + l16) * K + k0 + 16 * g);
                acc[n] = __builtin_amdgcn_wmma_f32_16x16x32_f16(
                    false, a, false, b, (short)0, acc[n], false, false);
            }
        }
        #pragma unroll
        for (int n = 0; n < NGRP; ++n) {
            const int ocol = (nb + n) * 16 + l16;
            const float bv = BIAS ? bias[ocol] : 0.0f;
            #pragma unroll
            for (int r = 0; r < 8; ++r) {
                const int orow = mt * 16 + r + 8 * g;
                float vv = acc[n][r] + bv;
                if (RES)  vv += residual[(size_t)orow * NC + ocol];
                if (RELU) vv = fmaxf(vv, 0.0f);
                O[(size_t)orow * NC + ocol] = vv;
            }
        }
    }
}

// ---------------------------------------------------------------------------
// Edge-bias WMMA GEMM: ebias[E,8] = edge_feat[E,64] @ Wpad[16,64]^T (+be)
// Wpad rows 8..15 are zero. One wave per 16-edge tile, K=64 (2 wmma).
// edge_feat is the dominant HBM stream (164 MB) -> read once + prefetch.
// ---------------------------------------------------------------------------
__global__ __launch_bounds__(256)
void ebias_gemm(const float* __restrict__ ef, const _Float16* __restrict__ Wpad,
                const float* __restrict__ be, float* __restrict__ ebias)
{
    const int lane  = threadIdx.x & 31;
    const int wave  = (blockIdx.x * blockDim.x + threadIdx.x) >> 5;
    const int nwave = (gridDim.x * blockDim.x) >> 5;
    const int tiles = EE >> 4;
    const int l16 = lane & 15;
    const int g   = lane >> 4;

    for (int t = wave; t < tiles; t += nwave) {
        const int arow = t * 16 + l16;          // edge index for A
        if (t + nwave < tiles)                  // prefetch next tile's stream
            __builtin_prefetch(ef + (size_t)(t + nwave) * 16 * EDIM + l16 * EDIM, 0, 1);
        v8f acc = {};
        #pragma unroll
        for (int k0 = 0; k0 < EDIM; k0 += 32) {
            v16h a;
            const float4* xa = (const float4*)(ef + (size_t)arow * EDIM + k0 + 16 * g);
            float4 t0 = xa[0], t1 = xa[1], t2 = xa[2], t3 = xa[3];
            a[0]=(_Float16)t0.x; a[1]=(_Float16)t0.y; a[2]=(_Float16)t0.z; a[3]=(_Float16)t0.w;
            a[4]=(_Float16)t1.x; a[5]=(_Float16)t1.y; a[6]=(_Float16)t1.z; a[7]=(_Float16)t1.w;
            a[8]=(_Float16)t2.x; a[9]=(_Float16)t2.y; a[10]=(_Float16)t2.z; a[11]=(_Float16)t2.w;
            a[12]=(_Float16)t3.x; a[13]=(_Float16)t3.y; a[14]=(_Float16)t3.z; a[15]=(_Float16)t3.w;
            const v16h b = *(const v16h*)(Wpad + (size_t)l16 * EDIM + k0 + 16 * g);
            acc = __builtin_amdgcn_wmma_f32_16x16x32_f16(
                false, a, false, b, (short)0, acc, false, false);
        }
        if (l16 < HH) {                          // only columns 0..7 are real heads
            const float bv = be[l16];
            #pragma unroll
            for (int r = 0; r < 8; ++r) {
                const int e = t * 16 + r + 8 * g;
                ebias[(size_t)e * HH + l16] = acc[r] + bv;
            }
        }
    }
}

// ---------------------------------------------------------------------------
__global__ void zero_f32(float* p, int n)
{
    int i = blockIdx.x * blockDim.x + threadIdx.x;
    if (i < n) p[i] = 0.0f;
}

__global__ void f32_to_f16(const float* __restrict__ s, _Float16* __restrict__ d, int n)
{
    int i = blockIdx.x * blockDim.x + threadIdx.x;
    if (i < n) d[i] = (_Float16)s[i];
}

// Wpad[16,64]: rows<8 = We, rows>=8 = 0
__global__ void build_wpad(const float* __restrict__ We, _Float16* __restrict__ Wpad)
{
    int i = blockIdx.x * blockDim.x + threadIdx.x;
    if (i < 16 * EDIM) Wpad[i] = (i < HH * EDIM) ? (_Float16)We[i] : (_Float16)0.0f;
}

// monotone float <-> uint mapping for atomic max over floats
__device__ __forceinline__ unsigned fkey(float f)
{
    unsigned u = __float_as_uint(f);
    return (u & 0x80000000u) ? ~u : (u | 0x80000000u);
}
__device__ __forceinline__ float funkey(unsigned k)
{
    unsigned u = (k & 0x80000000u) ? (k ^ 0x80000000u) : ~k;
    return __uint_as_float(u);
}

// ---------------------------------------------------------------------------
// scores[e,h] = dot(qd[dst], kp[src])/4 + ebias(already in scr); segment max.
// ---------------------------------------------------------------------------
__global__ void edge_scores(const float* __restrict__ qd, const float* __restrict__ kp,
                            const int* __restrict__ src, const int* __restrict__ dst,
                            float* __restrict__ scr, unsigned* __restrict__ smax)
{
    int idx = blockIdx.x * blockDim.x + threadIdx.x;
    if (idx >= EE * HH) return;
    int e = idx >> 3, h = idx & 7;
    int s = src[e], d = dst[e];
    const float4* qv = (const float4*)(qd + (size_t)d * CC + h * DD);
    const float4* kv = (const float4*)(kp + (size_t)s * CC + h * DD);
    float acc = 0.0f;
    #pragma unroll
    for (int i = 0; i < 4; ++i) {
        float4 a = qv[i], b = kv[i];
        acc += a.x * b.x + a.y * b.y + a.z * b.z + a.w * b.w;
    }
    acc = acc * 0.25f + scr[idx];   // 1/sqrt(16) + edge bias
    scr[idx] = acc;
    atomicMax(&smax[(size_t)d * HH + h], fkey(acc));
}

__global__ void edge_exp(float* __restrict__ scores, const unsigned* __restrict__ smax,
                         const int* __restrict__ dst, float* __restrict__ denom)
{
    int idx = blockIdx.x * blockDim.x + threadIdx.x;
    if (idx >= EE * HH) return;
    int e = idx >> 3, h = idx & 7;
    int d = dst[e];
    float m  = funkey(smax[(size_t)d * HH + h]);
    float ex = __expf(scores[idx] - m);
    scores[idx] = ex;
    atomicAdd(&denom[(size_t)d * HH + h], ex);
}

__global__ void edge_agg(const float* __restrict__ ex, const float* __restrict__ denom,
                         const float* __restrict__ vp,
                         const int* __restrict__ src, const int* __restrict__ dst,
                         float* __restrict__ agg)
{
    int idx = blockIdx.x * blockDim.x + threadIdx.x;
    if (idx >= EE * HH) return;
    int e = idx >> 3, h = idx & 7;
    int s = src[e], d = dst[e];
    float a = ex[idx] / denom[(size_t)d * HH + h];
    const float* vrow = vp  + (size_t)s * CC + h * DD;
    float*       arow = agg + (size_t)d * CC + h * DD;
    #pragma unroll
    for (int i = 0; i < DD; ++i)
        atomicAdd(&arow[i], a * vrow[i]);
}

// ---------------------------------------------------------------------------
__global__ void bn_stats(const float* __restrict__ x, float* __restrict__ sum,
                         float* __restrict__ sumsq, int n, int c)
{
    int col = threadIdx.x;
    float s = 0.0f, ss = 0.0f;
    for (int r = blockIdx.x; r < n; r += gridDim.x) {
        float v = x[(size_t)r * c + col];
        s += v; ss += v * v;
    }
    atomicAdd(&sum[col], s);
    atomicAdd(&sumsq[col], ss);
}

__global__ void bn_finalize(const float* __restrict__ sum, const float* __restrict__ sumsq,
                            const float* __restrict__ gamma, const float* __restrict__ beta,
                            float* __restrict__ scale, float* __restrict__ shift, int n)
{
    int c = threadIdx.x;
    float mu  = sum[c] / (float)n;
    float var = sumsq[c] / (float)n - mu * mu;
    float sc  = gamma[c] * rsqrtf(var + EPSV);
    scale[c] = sc;
    shift[c] = beta[c] - mu * sc;
}

__global__ void bn_apply(const float* __restrict__ x, const float* __restrict__ scale,
                         const float* __restrict__ shift, float* __restrict__ y,
                         int total, int c)
{
    int i = blockIdx.x * blockDim.x + threadIdx.x;
    if (i >= total) return;
    int col = i & (c - 1);
    y[i] = x[i] * scale[col] + shift[col];
}

// ---------------------------------------------------------------------------
extern "C" void kernel_launch(void* const* d_in, const int* in_sizes, int n_in,
                              void* d_out, int out_size, void* d_ws, size_t ws_size,
                              hipStream_t stream)
{
    const float* q  = (const float*)d_in[0];
    const float* k  = (const float*)d_in[1];
    const float* v  = (const float*)d_in[2];
    const float* ef = (const float*)d_in[3];
    const int*  src = (const int*)d_in[4];
    const int*  dst = (const int*)d_in[5];
    const float* Wq = (const float*)d_in[6];
    const float* Wk = (const float*)d_in[7];
    const float* Wv = (const float*)d_in[8];
    const float* We = (const float*)d_in[9];
    const float* be = (const float*)d_in[10];
    const float* Wo = (const float*)d_in[11];
    const float* W1 = (const float*)d_in[12];
    const float* b1 = (const float*)d_in[13];
    const float* W2 = (const float*)d_in[14];
    const float* b2 = (const float*)d_in[15];
    const float* g1 = (const float*)d_in[16];
    const float* bt1= (const float*)d_in[17];
    const float* g2 = (const float*)d_in[18];
    const float* bt2= (const float*)d_in[19];
    float* out = (float*)d_out;

    // ---- workspace carving ----
    const size_t NCf = (size_t)NN * CC;    // 5,120,000 floats
    const size_t NHf = (size_t)NN * HH;    // 320,000 floats
    float* ws   = (float*)d_ws;
    float* qd   = ws + 0 * NCf;            // qd, later rst
    float* kp   = ws + 1 * NCf;            // k proj, later FFN hidden [N,256]
    float* vp   = ws + 2 * NCf;            //   (kp..vp contiguous)
    float* agg  = ws + 3 * NCf;            // agg, later y
    float* scr  = ws + 4 * NCf;            // ebias -> scores -> ex (E*H floats)
    float* den  = ws + 5 * NCf;
    unsigned* smax = (unsigned*)(ws + 5 * NCf + NHf);
    float* stats = ws + 5 * NCf + 2 * NHf; // 8 x 128 floats
    float *sumA = stats,       *sqA = stats + 128;
    float *scA  = stats + 256, *shA = stats + 384;
    float *sumB = stats + 512, *sqB = stats + 640;
    float *scB  = stats + 768, *shB = stats + 896;
    // f16 weight copies (converted once per launch)
    _Float16* Wq16 = (_Float16*)(stats + 1024);
    _Float16* Wk16 = Wq16 + 16384;
    _Float16* Wv16 = Wk16 + 16384;
    _Float16* Wo16 = Wv16 + 16384;
    _Float16* W116 = Wo16 + 16384;   // 256*128
    _Float16* W216 = W116 + 32768;   // 128*256
    _Float16* Wp16 = W216 + 32768;   // 16*64
    float* rst    = qd;
    float* hidden = kp;
    float* yb     = agg;

    const int TB = 256;

    // ---- weight conversion + init ----
    f32_to_f16<<<64, TB, 0, stream>>>(Wq, Wq16, 16384);
    f32_to_f16<<<64, TB, 0, stream>>>(Wk, Wk16, 16384);
    f32_to_f16<<<64, TB, 0, stream>>>(Wv, Wv16, 16384);
    f32_to_f16<<<64, TB, 0, stream>>>(Wo, Wo16, 16384);
    f32_to_f16<<<128, TB, 0, stream>>>(W1, W116, 32768);
    f32_to_f16<<<128, TB, 0, stream>>>(W2, W216, 32768);
    build_wpad<<<4, TB, 0, stream>>>(We, Wp16);
    zero_f32<<<((int)NCf + TB - 1) / TB, TB, 0, stream>>>(agg, (int)NCf);
    zero_f32<<<((int)NHf + TB - 1) / TB, TB, 0, stream>>>(den, (int)NHf);
    zero_f32<<<((int)NHf + TB - 1) / TB, TB, 0, stream>>>((float*)smax, (int)NHf);
    zero_f32<<<4, TB, 0, stream>>>(stats, 1024);

    // ---- projections: [40000,128] @ [128,128]^T ----
    gemm_wmma<128, 4, false, false, false><<<625, TB, 0, stream>>>(q, Wq16, nullptr, nullptr, qd, NN, CC);
    gemm_wmma<128, 4, false, false, false><<<625, TB, 0, stream>>>(k, Wk16, nullptr, nullptr, kp, NN, CC);
    gemm_wmma<128, 4, false, false, false><<<625, TB, 0, stream>>>(v, Wv16, nullptr, nullptr, vp, NN, CC);

    // ---- edge bias GEMM (streams edge_feat once) ----
    ebias_gemm<<<5000, TB, 0, stream>>>(ef, Wp16, be, scr);

    // ---- edge phase ----
    const int EHB = (EE * HH + TB - 1) / TB;
    edge_scores<<<EHB, TB, 0, stream>>>(qd, kp, src, dst, scr, smax);
    edge_exp  <<<EHB, TB, 0, stream>>>(scr, smax, dst, den);
    edge_agg  <<<EHB, TB, 0, stream>>>(scr, den, vp, src, dst, agg);

    // ---- output proj + residual, BN1 ----
    gemm_wmma<128, 4, false, true, false><<<625, TB, 0, stream>>>(agg, Wo16, nullptr, q, rst, NN, CC);
    bn_stats   <<<512, 128, 0, stream>>>(rst, sumA, sqA, NN, CC);
    bn_finalize<<<1, 128, 0, stream>>>(sumA, sqA, g1, bt1, scA, shA, NN);
    bn_apply   <<<((int)NCf + TB - 1) / TB, TB, 0, stream>>>(rst, scA, shA, rst, (int)NCf, CC);

    // ---- FFN ----
    gemm_wmma<128, 4, true, false, true ><<<1250, TB, 0, stream>>>(rst, W116, b1, nullptr, hidden, NN, 2 * CC);
    gemm_wmma<256, 4, true, true,  false><<<625,  TB, 0, stream>>>(hidden, W216, b2, rst, yb, NN, CC);

    // ---- BN2 -> out ----
    bn_stats   <<<512, 128, 0, stream>>>(yb, sumB, sqB, NN, CC);
    bn_finalize<<<1, 128, 0, stream>>>(sumB, sqB, g2, bt2, scB, shB, NN);
    bn_apply   <<<((int)NCf + TB - 1) / TB, TB, 0, stream>>>(yb, scB, shB, out, (int)NCf, CC);
}
